// MultiConvGRU_26903675142710
// MI455X (gfx1250) — compile-verified, matrix operand order checked by reading, not run on hardware
//
#include <hip/hip_runtime.h>
#include <hip/hip_bf16.h>

// ---------------------------------------------------------------------------
// MultiConvGRU on gfx1250 (MI455X). LoRA convs fold exactly into effective
// 3x3 convs (W_eff = W + 2*A·Bl). Convs run as implicit GEMM with
// v_wmma_f32_16x16x32_bf16 (f32 accumulate).
//
// K-dimension is PERMUTED to k' = (kh*3+kw)*CIN + ci; the LDS input patch is
// channel-last, so each lane's A fragment is two ds_load_b128 at immediate
// offsets. Packed weights are copied to LDS once per block, so B fragments
// are also ds_load_b128 at immediate offsets. A/B fragments are explicitly
// double-buffered in registers: iteration kc+1's DS loads are issued before
// iteration kc's WMMAs so LDS latency overlaps matrix math.
// ---------------------------------------------------------------------------

#define NB 8
#define NT 12
#define CO 64
#define HWD 64
#define IMG_PIX (HWD * HWD)      // 4096
#define CHW (CO * IMG_PIX)       // 262144 floats per 64-ch image

typedef __attribute__((ext_vector_type(16))) __bf16 v16bf;
typedef __attribute__((ext_vector_type(8)))  __bf16 v8bf;
typedef __attribute__((ext_vector_type(8)))  float  v8f;

// ---------------------------------------------------------------------------
// Fold LoRA into effective 3x3 weights, pack into WMMA B-fragment order with
// permuted K:  k' = (kh*3+kw)*cin + ci.
// Packed layout: wpack[(((nt*KC + kc)*32 + lane)*16 + j)]
//   cout = nt*16 + (lane&15)
//   k'   = kc*32 + (j<8 ? 0 : 16) + (lane>=16 ? 8 : 0) + (j&7)
// ---------------------------------------------------------------------------
__global__ void pack_weights(const float* __restrict__ W,
                             const float* __restrict__ A,
                             const float* __restrict__ Bl,
                             __bf16* __restrict__ wpack,
                             int cin, int KC)
{
    const int K = cin * 9;
    const int total = 4 * KC * 32 * 16;
    for (int idx = blockIdx.x * blockDim.x + threadIdx.x; idx < total;
         idx += gridDim.x * blockDim.x) {
        const int j    = idx & 15;
        const int lane = (idx >> 4) & 31;
        const int rest = idx >> 9;
        const int kc   = rest % KC;
        const int nt   = rest / KC;
        const int o  = nt * 16 + (lane & 15);
        const int kp = kc * 32 + ((j < 8) ? 0 : 16) + ((lane >> 4) << 3) + (j & 7);
        const int p  = kp / cin;            // tap index kh*3+kw
        const int ci = kp - p * cin;
        float v = 0.0f;
        if (p < 9) {
            const int korig = ci * 9 + p;   // original [cin][3][3] flat index
            v = W[(size_t)o * K + korig];
            float lo = 0.0f;
            #pragma unroll
            for (int r = 0; r < 4; ++r)
                lo += A[o * 4 + r] * Bl[(size_t)r * K + korig];
            v += 2.0f * lo;                  // SCALE = ALPHA/R = 2
        }
        wpack[idx] = (__bf16)v;
    }
}

// ---------------------------------------------------------------------------
// Implicit-GEMM 3x3 SAME conv: M = pixels, N = cout (64), K' = 9*cin.
// Block: 256 threads (8 waves). Tile: 128 pixels x 64 couts.
// Dynamic LDS: [ input patch 4*66*CIN bf16 | packed weights 4*KC*512 bf16 ]
// ---------------------------------------------------------------------------
template <int CIN>
__global__ __launch_bounds__(256)
void conv3x3_wmma(const float* __restrict__ in, long in_img_stride,
                  const __bf16* __restrict__ wpack,
                  const float* __restrict__ bias,
                  float* __restrict__ out, long out_img_stride)
{
    constexpr int K      = CIN * 9;
    constexpr int KC     = (K + 31) / 32;
    constexpr int IN_ELT = 4 * 66 * CIN;        // halo patch elements
    constexpr int W_ELT  = 4 * KC * 32 * 16;    // packed weight elements

    extern __shared__ __bf16 smem[];
    __bf16* sIn = smem;                 // [r][xi][ci] channel-last, xi = x+1
    __bf16* sW  = smem + IN_ELT;        // LDS copy of wpack

    const int img  = blockIdx.y;
    const int tile = blockIdx.x;        // 0..31 (2 image rows each)
    const int tid  = threadIdx.x;
    const int wave = tid >> 5;
    const int lane = tid & 31;
    const int y0   = tile * 2;

    const float* inb = in + (long)img * in_img_stride;

    // Stage weights: straight contiguous copy (bf16 already packed), b128 ops.
    {
        const uint4* src = (const uint4*)wpack;
        uint4*       dst = (uint4*)sW;
        #pragma unroll
        for (int i = tid; i < W_ELT / 8; i += 256) dst[i] = src[i];
    }

    // Stage input halo patch: coalesced global reads, channel-last LDS writes.
    for (int idx = tid; idx < IN_ELT; idx += 256) {
        const int ci  = idx / (4 * 66);
        const int rem = idx - ci * (4 * 66);
        const int r   = rem / 66;
        const int xi  = rem - r * 66;   // 0..65 -> x = xi-1
        const int y   = y0 - 1 + r;
        const int xx  = xi - 1;
        float v = 0.0f;
        if (y >= 0 && y < HWD && xx >= 0 && xx < HWD)
            v = inb[ci * IMG_PIX + y * HWD + xx];
        sIn[(r * 66 + xi) * CIN + ci] = (__bf16)v;
    }
    __syncthreads();

    const int py  = y0 + (wave >> 2);   // this wave's image row
    const int pxb = (wave & 3) * 16;    // this wave's col base
    const int row = lane & 15;          // M row (A) / N col (B,C,D)
    const int hi  = lane >> 4;

    // A-side base: all runtime math folds here; per-run offsets are immediates.
    const __bf16* sBase = sIn + ((py - y0) * 66 + pxb + row) * CIN + hi * 8;
    // B-side bases: one per N-tile so every ds offset fits in 16-bit imm.
    const __bf16* sWb0 = sW + (size_t)lane * 16;
    const __bf16* sWb1 = sWb0 + (size_t)KC * 32 * 16;
    const __bf16* sWb2 = sWb1 + (size_t)KC * 32 * 16;
    const __bf16* sWb3 = sWb2 + (size_t)KC * 32 * 16;

    // Preload bias into f32 accumulators (bias depends on N = lane&15 only).
    v8f acc[4];
    #pragma unroll
    for (int nt = 0; nt < 4; ++nt) {
        const float bv = bias[nt * 16 + row];
        #pragma unroll
        for (int i = 0; i < 8; ++i) acc[nt][i] = bv;
    }

    union AU { v16bf v; v8bf h[2]; };

    auto loadA = [&](int kc, AU& dst) {
        #pragma unroll
        for (int run = 0; run < 2; ++run) {
            const int k0 = kc * 32 + run * 16;   // run base (sans hi*8)
            const int p  = k0 / CIN;             // tap: constant across hi
            if (p < 9) {
                const int kh  = p / 3;
                const int kw  = p - kh * 3;
                const int off = (kh * 66 + kw) * CIN + (k0 % CIN);
                dst.h[run] = *(const v8bf*)(sBase + off);
            } else {
                #pragma unroll
                for (int jj = 0; jj < 8; ++jj) dst.h[run][jj] = (__bf16)0.0f;
            }
        }
    };
    auto loadB = [&](int kc, v16bf* dst) {
        const size_t o = (size_t)kc * 32 * 16;
        dst[0] = *(const v16bf*)(sWb0 + o);
        dst[1] = *(const v16bf*)(sWb1 + o);
        dst[2] = *(const v16bf*)(sWb2 + o);
        dst[3] = *(const v16bf*)(sWb3 + o);
    };

    // Software-pipelined K loop: prefetch kc+1 fragments before kc's WMMAs.
    AU    a[2];
    v16bf bf[2][4];
    loadA(0, a[0]);
    loadB(0, bf[0]);
    #pragma unroll
    for (int kc = 0; kc < KC; ++kc) {
        const int cur = kc & 1;
        const int nxt = cur ^ 1;
        if (kc + 1 < KC) {
            loadA(kc + 1, a[nxt]);
            loadB(kc + 1, bf[nxt]);
        }
        #pragma unroll
        for (int nt = 0; nt < 4; ++nt) {
            acc[nt] = __builtin_amdgcn_wmma_f32_16x16x32_bf16(
                false, a[cur].v, false, bf[cur][nt], (short)0, acc[nt],
                false, false);
        }
    }

    // Epilogue: lane holds N=lane&15; VGPR i holds M = i + hi*8 -> 8
    // consecutive pixels -> two b128 stores per N-tile.
    float* outb = out + (long)img * out_img_stride;
    #pragma unroll
    for (int nt = 0; nt < 4; ++nt) {
        const int co = nt * 16 + row;
        float* op = outb + (size_t)co * IMG_PIX + py * HWD + pxb + hi * 8;
        float4 s0 = make_float4(acc[nt][0], acc[nt][1], acc[nt][2], acc[nt][3]);
        float4 s1 = make_float4(acc[nt][4], acc[nt][5], acc[nt][6], acc[nt][7]);
        *(float4*)(op)     = s0;
        *(float4*)(op + 4) = s1;
    }
}

// ---------------------------------------------------------------------------
// Fused gate + state update for one timestep.
//   z = sig(hzc + uz + bz); r = sig(hrc + ur + br)
//   h~ = leaky(hhc + r*uh + bh, 0.2); h = (1-z)h~ + z h
// ---------------------------------------------------------------------------
__global__ __launch_bounds__(256)
void gate_update(const float* __restrict__ hzc, const float* __restrict__ hrc,
                 const float* __restrict__ hhc,
                 const float* __restrict__ uz, const float* __restrict__ ur,
                 const float* __restrict__ uh,
                 const float* __restrict__ bz, const float* __restrict__ br,
                 const float* __restrict__ bh,
                 float* __restrict__ h, float* __restrict__ outp,
                 int t, int n)
{
    const int i = blockIdx.x * blockDim.x + threadIdx.x;
    if (i >= n) return;
    const int  c   = (i >> 12) & 63;               // channel (stride 4096)
    const long b   = (long)i >> 18;                // batch (stride 262144)
    const long rem = (long)i & (long)(CHW - 1);
    const long ui  = (b * NT + t) * (long)CHW + rem;

    const float z  = 1.0f / (1.0f + __expf(-(hzc[i] + uz[ui] + bz[c])));
    const float r  = 1.0f / (1.0f + __expf(-(hrc[i] + ur[ui] + br[c])));
    const float hb = hhc[i] + r * uh[ui] + bh[c];
    const float h_ = hb > 0.0f ? hb : 0.2f * hb;
    const float hn = (1.0f - z) * h_ + z * h[i];
    h[i]     = hn;
    outp[ui] = hn;
}

// ---------------------------------------------------------------------------
extern "C" void kernel_launch(void* const* d_in, const int* in_sizes, int n_in,
                              void* d_out, int out_size, void* d_ws, size_t ws_size,
                              hipStream_t stream)
{
    (void)in_sizes; (void)n_in; (void)out_size; (void)ws_size;

    // Input order: x, then 12 LoRA param groups (W,b,A,Bl) in names order,
    // then b_z, b_r, b_h.
    const float* x = (const float*)d_in[0];
    const float *W[12], *bias[12], *Am[12], *Bl[12];
    for (int i = 0; i < 12; ++i) {
        W[i]    = (const float*)d_in[1 + 4 * i + 0];
        bias[i] = (const float*)d_in[1 + 4 * i + 1];
        Am[i]   = (const float*)d_in[1 + 4 * i + 2];
        Bl[i]   = (const float*)d_in[1 + 4 * i + 3];
    }
    const float* bz = (const float*)d_in[49];
    const float* br = (const float*)d_in[50];
    const float* bh = (const float*)d_in[51];

    // Workspace carve-out (~540 MB): packed weights, x-path products, temps.
    char* ws = (char*)d_ws;
    size_t off = 0;
    auto alloc = [&](size_t bytes) -> char* {
        off = (off + 255) & ~(size_t)255;
        char* p = ws + off;
        off += bytes;
        return p;
    };
    const int KC16 = 5, KC64 = 18;     // ceil(cin*9 / 32)
    __bf16* wpack[12];
    for (int i = 0; i < 12; ++i) {
        const int kc = (i == 0 || i == 4 || i == 8) ? KC16 : KC64;
        wpack[i] = (__bf16*)alloc((size_t)4 * kc * 32 * 16 * sizeof(__bf16));
    }
    const size_t seq_bytes = (size_t)NB * NT * CHW * sizeof(float);
    float* uz  = (float*)alloc(seq_bytes);
    float* ur  = (float*)alloc(seq_bytes);
    float* uh  = (float*)alloc(seq_bytes);
    float* tA  = (float*)alloc(seq_bytes);
    float* tB  = (float*)alloc(seq_bytes);
    const size_t st_bytes = (size_t)NB * CHW * sizeof(float);
    float* hzc = (float*)alloc(st_bytes);
    float* hrc = (float*)alloc(st_bytes);
    float* hhc = (float*)alloc(st_bytes);
    float* h   = (float*)alloc(st_bytes);

    // Dynamic LDS sizes: halo patch + full packed-weight copy.
    const size_t sh16 = (size_t)(4 * 66 * 16 + 4 * KC16 * 32 * 16) * sizeof(__bf16);
    const size_t sh64 = (size_t)(4 * 66 * 64 + 4 * KC64 * 32 * 16) * sizeof(__bf16);
    // gfx1250 allows up to 320KB LDS per workgroup; lift the dynamic cap.
    (void)hipFuncSetAttribute((const void*)&conv3x3_wmma<16>,
                              hipFuncAttributeMaxDynamicSharedMemorySize, (int)sh16);
    (void)hipFuncSetAttribute((const void*)&conv3x3_wmma<64>,
                              hipFuncAttributeMaxDynamicSharedMemorySize, (int)sh64);

    // 1) Fold LoRA -> effective bf16 weights in permuted B-fragment order.
    for (int i = 0; i < 12; ++i) {
        const int cin = (i == 0 || i == 4 || i == 8) ? 16 : 64;
        const int kc  = (cin == 16) ? KC16 : KC64;
        const int total = 4 * kc * 32 * 16;
        pack_weights<<<(total + 255) / 256, 256, 0, stream>>>(
            W[i], Am[i], Bl[i], wpack[i], cin, kc);
    }

    // 2) h0 = 0
    hipMemsetAsync(h, 0, st_bytes, stream);

    // 3) x-path chains for all (b,t) in parallel: xz*, xr*, xh*.
    const dim3 blk(256);
    const dim3 gx(32, NB * NT);
    const long xstride = (long)16 * IMG_PIX;
    conv3x3_wmma<16><<<gx, blk, sh16, stream>>>(x,  xstride, wpack[0],  bias[0],  tA, (long)CHW);
    conv3x3_wmma<64><<<gx, blk, sh64, stream>>>(tA, (long)CHW, wpack[1],  bias[1],  tB, (long)CHW);
    conv3x3_wmma<64><<<gx, blk, sh64, stream>>>(tB, (long)CHW, wpack[2],  bias[2],  uz, (long)CHW);

    conv3x3_wmma<16><<<gx, blk, sh16, stream>>>(x,  xstride, wpack[4],  bias[4],  tA, (long)CHW);
    conv3x3_wmma<64><<<gx, blk, sh64, stream>>>(tA, (long)CHW, wpack[5],  bias[5],  tB, (long)CHW);
    conv3x3_wmma<64><<<gx, blk, sh64, stream>>>(tB, (long)CHW, wpack[6],  bias[6],  ur, (long)CHW);

    conv3x3_wmma<16><<<gx, blk, sh16, stream>>>(x,  xstride, wpack[8],  bias[8],  tA, (long)CHW);
    conv3x3_wmma<64><<<gx, blk, sh64, stream>>>(tA, (long)CHW, wpack[9],  bias[9],  tB, (long)CHW);
    conv3x3_wmma<64><<<gx, blk, sh64, stream>>>(tB, (long)CHW, wpack[10], bias[10], uh, (long)CHW);

    // 4) Sequential recurrence: 3 state convs + fused gate per timestep.
    const dim3 gh(32, NB);
    const int n = NB * CHW;
    float* outp = (float*)d_out;
    for (int t = 0; t < NT; ++t) {
        conv3x3_wmma<64><<<gh, blk, sh64, stream>>>(h, (long)CHW, wpack[3],  bias[3],  hzc, (long)CHW);
        conv3x3_wmma<64><<<gh, blk, sh64, stream>>>(h, (long)CHW, wpack[7],  bias[7],  hrc, (long)CHW);
        conv3x3_wmma<64><<<gh, blk, sh64, stream>>>(h, (long)CHW, wpack[11], bias[11], hhc, (long)CHW);
        gate_update<<<(n + 255) / 256, 256, 0, stream>>>(
            hzc, hrc, hhc, uz, ur, uh, bz, br, bh, h, outp, t, n);
    }
}
